// GraphGRU_33011118637674
// MI455X (gfx1250) — compile-verified, matrix operand order checked by reading, not run on hardware
//
#include <hip/hip_runtime.h>
#include <hip/hip_bf16.h>
#include <math.h>

#define BB 8
#define TT 4
#define NN 4096
#define F_IN 3
#define UU 64
#define D_OUT 3
#define EE 65536
#define EF (EE + NN)          // 69632 edges incl. self loops
#define ROWS (BB * NN)        // 32768
#define KP 96                 // K padded (67 -> 96, three K=32 WMMA chunks)
#define NEG_SLOPE 0.2f

typedef __attribute__((ext_vector_type(16))) _Float16 v16h;
typedef __attribute__((ext_vector_type(8)))  float    v8f;

__device__ __forceinline__ v8f wmma_f16(v16h a, v16h b, v8f c) {
  return __builtin_amdgcn_wmma_f32_16x16x32_f16(false, a, false, b, (short)0, c,
                                                false, false);
}

// K index held by element i of a v16h A/B fragment for this lane.
// lanes 0-15: K in {0..7, 16..23}; lanes 16-31: K in {8..15, 24..31}.
__device__ __forceinline__ int frag_k(int i, int lane) {
  int kb = (lane & 16) ? 8 : 0;
  return kb + i + ((i >> 3) << 3);
}

// Build an fp16 A/B fragment from a row whose K-run [kb.., kb+16..] is
// contiguous fp32 (works for global or LDS pointers).
__device__ __forceinline__ v16h load_a96(const float* p) {
  float4 f0 = *(const float4*)(p + 0);
  float4 f1 = *(const float4*)(p + 4);
  float4 f2 = *(const float4*)(p + 16);
  float4 f3 = *(const float4*)(p + 20);
  v16h a;
  a[0] = (_Float16)f0.x;  a[1] = (_Float16)f0.y;
  a[2] = (_Float16)f0.z;  a[3] = (_Float16)f0.w;
  a[4] = (_Float16)f1.x;  a[5] = (_Float16)f1.y;
  a[6] = (_Float16)f1.z;  a[7] = (_Float16)f1.w;
  a[8] = (_Float16)f2.x;  a[9] = (_Float16)f2.y;
  a[10] = (_Float16)f2.z; a[11] = (_Float16)f2.w;
  a[12] = (_Float16)f3.x; a[13] = (_Float16)f3.y;
  a[14] = (_Float16)f3.z; a[15] = (_Float16)f3.w;
  return a;
}

__device__ __forceinline__ void atomicMaxF(float* addr, float v) {
  if (v >= 0.0f) atomicMax((int*)addr, __float_as_int(v));
  else           atomicMin((unsigned int*)addr, __float_as_uint(v));
}

__global__ void fill_kernel(float* __restrict__ p, float v, int n) {
  int i = blockIdx.x * blockDim.x + threadIdx.x;
  if (i < n) p[i] = v;
}

// Pack a (67 x ldn) fp32 weight into fragment-ordered fp16:
// dst[((kcI*NT + nt)*32 + lane)*16 + i] = W[k*ldn + nt*16 + (lane&15)]
__global__ void pack_w_kernel(const float* __restrict__ W, int ldn, int NT,
                              _Float16* __restrict__ dst) {
  int gid = blockIdx.x * blockDim.x + threadIdx.x;
  int total = 3 * NT * 512;
  if (gid >= total) return;
  int i    = gid & 15;
  int lane = (gid >> 4) & 31;
  int rest = gid >> 9;
  int nt   = rest % NT;
  int kcI  = rest / NT;
  int k = kcI * 32 + frag_k(i, lane);
  int n = nt * 16 + (lane & 15);
  dst[gid] = (_Float16)((k < UU + F_IN) ? W[k * ldn + n] : 0.0f);
}

// cbuf (ROWS x 96): [x_t(3) | s = agg + gat_b (64) | zero pad (29)]
__global__ void build_conc_kernel(const float* __restrict__ x,
                                  const float* __restrict__ agg,
                                  const float* __restrict__ gat_b,
                                  float* __restrict__ cbuf, int t) {
  int gid = blockIdx.x * blockDim.x + threadIdx.x;
  if (gid >= ROWS * KP) return;
  int col = gid % KP, r = gid / KP;
  float v = 0.0f;
  if (col < F_IN) {
    int b = r >> 12, n = r & (NN - 1);
    v = x[((size_t)((b * TT + t) * NN + n)) * F_IN + col];
  } else if (col < F_IN + UU) {
    int j = col - F_IN;
    v = agg[(size_t)r * UU + j] + gat_b[j];
  }
  cbuf[gid] = v;
}

// ---------------------------------------------------------------------------
// hp = [h, x_t] @ gat_W   (ROWS x 67) x (67 x 64). Wave per 16-row tile,
// 4 n-tiles x 3 K chunks = 12 WMMA / wave. B fragments from packed fp16.
// ---------------------------------------------------------------------------
__global__ __launch_bounds__(128)
void gat_gemm_kernel(const float* __restrict__ h,
                     const float* __restrict__ x,
                     const _Float16* __restrict__ Wf,  // packed, NT=4
                     float* __restrict__ hp, int t) {
  const int lane  = threadIdx.x & 31;
  const int wave  = threadIdx.x >> 5;
  const int tile  = blockIdx.x * 4 + wave;
  const int row0  = tile * 16;
  const int rlane = lane & 15;
  const int kb    = (lane & 16) ? 8 : 0;
  const int grow  = row0 + rlane;
  const int b     = grow >> 12;
  const int n_id  = grow & (NN - 1);
  const float* xrow = x + ((size_t)((b * TT + t) * NN + n_id)) * F_IN;
  __builtin_prefetch(Wf + (size_t)lane * 16, 0, 0);

  v8f acc[4] = {};
#pragma unroll
  for (int kcI = 0; kcI < 3; ++kcI) {
    v16h a;
    if (kcI < 2) {                       // K chunk fully inside h (cols 0..63)
      a = load_a96(h + (size_t)grow * UU + kcI * 32 + kb);
    } else {                             // tail: x features at k = 64..66
      a = (v16h){};
      if (kb == 0) {
        a[0] = (_Float16)xrow[0];
        a[1] = (_Float16)xrow[1];
        a[2] = (_Float16)xrow[2];
      }
    }
#pragma unroll
    for (int nt = 0; nt < 4; ++nt) {
      v16h bf = *(const v16h*)(Wf + ((size_t)(kcI * 4 + nt) * 32 + lane) * 16);
      acc[nt] = wmma_f16(a, bf, acc[nt]);
    }
  }
  const int mofs = (lane & 16) ? 8 : 0;
#pragma unroll
  for (int nt = 0; nt < 4; ++nt) {
    int n = nt * 16 + rlane;
#pragma unroll
    for (int vr = 0; vr < 8; ++vr)
      hp[(size_t)(row0 + vr + mofs) * UU + n] = acc[nt][vr];
  }
}

// es = hp . a_src, ed = hp . a_dst (per node)
__global__ void attn_node_kernel(const float* __restrict__ hp,
                                 const float* __restrict__ a_src,
                                 const float* __restrict__ a_dst,
                                 float* __restrict__ es,
                                 float* __restrict__ ed) {
  int r = blockIdx.x * blockDim.x + threadIdx.x;
  if (r >= ROWS) return;
  float s = 0.0f, d = 0.0f;
#pragma unroll 8
  for (int j = 0; j < UU; ++j) {
    float v = hp[(size_t)r * UU + j];
    s += v * a_src[j];
    d += v * a_dst[j];
  }
  es[r] = s;
  ed[r] = d;
}

// leaky_relu(es[src] + ed[dst]); segment max over dst (atomic)
__global__ void edge_logit_kernel(const float* __restrict__ es,
                                  const float* __restrict__ ed,
                                  const int* __restrict__ src,
                                  const int* __restrict__ dst,
                                  float* __restrict__ mbuf,
                                  float* __restrict__ lg) {
  int e = blockIdx.x * blockDim.x + threadIdx.x;
  if (e >= BB * EF) return;
  int b = e / EF, k = e % EF;
  int sN = (k < EE) ? src[k] : (k - EE);
  int dN = (k < EE) ? dst[k] : (k - EE);
  float v = es[b * NN + sN] + ed[b * NN + dN];
  v = (v > 0.0f) ? v : NEG_SLOPE * v;
  lg[e] = v;
  atomicMaxF(&mbuf[b * NN + dN], v);
}

// exp(logit - max); segment sum over dst (atomic)
__global__ void edge_exp_kernel(const int* __restrict__ dst,
                                const float* __restrict__ mbuf,
                                float* __restrict__ lg,
                                float* __restrict__ ssum) {
  int e = blockIdx.x * blockDim.x + threadIdx.x;
  if (e >= BB * EF) return;
  int b = e / EF, k = e % EF;
  int dN = (k < EE) ? dst[k] : (k - EE);
  float v = __expf(lg[e] - mbuf[b * NN + dN]);
  lg[e] = v;
  atomicAdd(&ssum[b * NN + dN], v);
}

// agg[dst] += alpha * hp[src]   (16 threads per edge, 4 features each)
__global__ void edge_agg_kernel(const int* __restrict__ src,
                                const int* __restrict__ dst,
                                const float* __restrict__ lg,
                                const float* __restrict__ ssum,
                                const float* __restrict__ hp,
                                float* __restrict__ agg) {
  int gid = blockIdx.x * blockDim.x + threadIdx.x;
  if (gid >= BB * EF * 16) return;
  int e = gid >> 4;
  int c4 = (gid & 15) * 4;
  int b = e / EF, k = e % EF;
  int sN = (k < EE) ? src[k] : (k - EE);
  int dN = (k < EE) ? dst[k] : (k - EE);
  float alpha = lg[e] / ssum[b * NN + dN];
  const float* hs = hp + (size_t)(b * NN + sN) * UU + c4;
  float* ag = agg + (size_t)(b * NN + dN) * UU + c4;
#pragma unroll
  for (int j = 0; j < 4; ++j) atomicAdd(&ag[j], hs[j] * alpha);
}

// ---------------------------------------------------------------------------
// Fused GRU cell, A1 = cbuf rows (96-wide, pre-concat + bias folded):
//   ru = sigmoid(cbuf @ W1 + b1); r,u = split(ru)
//   c  = tanh([x, r*s] @ W2 + b2)        (A2 staged 96-wide in LDS)
//   h  = u*s + (1-u)*c
// gru1: 8 n-tiles x 3 K = 24 WMMA; gru2: 4 x 3 = 12 WMMA per wave.
// ---------------------------------------------------------------------------
__global__ __launch_bounds__(128)
void gru_kernel(const float* __restrict__ cbuf,
                const _Float16* __restrict__ W1f,  // packed, NT=8
                const _Float16* __restrict__ W2f,  // packed, NT=4
                const float* __restrict__ b1,
                const float* __restrict__ b2,
                float* __restrict__ h) {
  __shared__ float rsb[4][16][KP];
  const int lane  = threadIdx.x & 31;
  const int wave  = threadIdx.x >> 5;
  const int tile  = blockIdx.x * 4 + wave;
  const int row0  = tile * 16;
  const int rlane = lane & 15;
  const int kb    = (lane & 16) ? 8 : 0;
  const int grow  = row0 + rlane;
  __builtin_prefetch(W1f + (size_t)lane * 16, 0, 0);

  // Pre-fill A2 stage: x in cols 0..2, zeros elsewhere (3..66 overwritten).
  for (int idx = lane; idx < 16 * KP; idx += 32) {
    int row = idx / KP, col = idx % KP;
    float v = (col < F_IN) ? cbuf[(size_t)(row0 + row) * KP + col] : 0.0f;
    rsb[wave][row][col] = v;
  }

  // ---- gru1: cbuf @ W1 (67 x 128) ----
  v8f acc1[8] = {};
#pragma unroll
  for (int kcI = 0; kcI < 3; ++kcI) {
    v16h a = load_a96(cbuf + (size_t)grow * KP + kcI * 32 + kb);
#pragma unroll
    for (int nt = 0; nt < 8; ++nt) {
      v16h bf = *(const v16h*)(W1f + ((size_t)(kcI * 8 + nt) * 32 + lane) * 16);
      acc1[nt] = wmma_f16(a, bf, acc1[nt]);
    }
  }

  const int mofs = (lane & 16) ? 8 : 0;
  float ureg[4][8];
#pragma unroll
  for (int nt = 0; nt < 4; ++nt) {
    int n = nt * 16 + rlane;
#pragma unroll
    for (int vr = 0; vr < 8; ++vr) {
      int mm = vr + mofs;
      float rr = 1.0f / (1.0f + __expf(-(acc1[nt][vr] + b1[n])));
      float sv = cbuf[(size_t)(row0 + mm) * KP + F_IN + n];   // s (bias folded)
      rsb[wave][mm][F_IN + n] = rr * sv;                      // r * s
      ureg[nt][vr] = 1.0f / (1.0f + __expf(-(acc1[nt + 4][vr] + b1[UU + n])));
    }
  }
  __syncthreads();

  // ---- gru2: [x, r*s] @ W2 (67 x 64) ----
  v8f acc2[4] = {};
#pragma unroll
  for (int kcI = 0; kcI < 3; ++kcI) {
    v16h a = load_a96(&rsb[wave][rlane][kcI * 32 + kb]);
#pragma unroll
    for (int nt = 0; nt < 4; ++nt) {
      v16h bf = *(const v16h*)(W2f + ((size_t)(kcI * 4 + nt) * 32 + lane) * 16);
      acc2[nt] = wmma_f16(a, bf, acc2[nt]);
    }
  }

#pragma unroll
  for (int nt = 0; nt < 4; ++nt) {
    int n = nt * 16 + rlane;
#pragma unroll
    for (int vr = 0; vr < 8; ++vr) {
      int mm = vr + mofs;
      float c  = tanhf(acc2[nt][vr] + b2[n]);
      float sv = cbuf[(size_t)(row0 + mm) * KP + F_IN + n];
      float uu = ureg[nt][vr];
      h[(size_t)(row0 + mm) * UU + n] = uu * sv + (1.0f - uu) * c;
    }
  }
}

// out = h_final @ out_W (64 x 3) + out_b
__global__ void out_kernel(const float* __restrict__ h,
                           const float* __restrict__ Wo,
                           const float* __restrict__ bo,
                           float* __restrict__ out) {
  int r = blockIdx.x * blockDim.x + threadIdx.x;
  if (r >= ROWS) return;
  float a0 = bo[0], a1 = bo[1], a2 = bo[2];
#pragma unroll 8
  for (int j = 0; j < UU; ++j) {
    float v = h[(size_t)r * UU + j];
    a0 += v * Wo[j * D_OUT + 0];
    a1 += v * Wo[j * D_OUT + 1];
    a2 += v * Wo[j * D_OUT + 2];
  }
  out[r * D_OUT + 0] = a0;
  out[r * D_OUT + 1] = a1;
  out[r * D_OUT + 2] = a2;
}

extern "C" void kernel_launch(void* const* d_in, const int* in_sizes, int n_in,
                              void* d_out, int out_size, void* d_ws, size_t ws_size,
                              hipStream_t stream) {
  const float* x      = (const float*)d_in[0];
  const int*   src    = (const int*)  d_in[1];
  const int*   dst    = (const int*)  d_in[2];
  const float* gat_W  = (const float*)d_in[3];
  const float* a_src  = (const float*)d_in[4];
  const float* a_dst  = (const float*)d_in[5];
  const float* gat_b  = (const float*)d_in[6];
  const float* gru1_W = (const float*)d_in[7];
  const float* gru1_b = (const float*)d_in[8];
  const float* gru2_W = (const float*)d_in[9];
  const float* gru2_b = (const float*)d_in[10];
  const float* out_W  = (const float*)d_in[11];
  const float* out_b  = (const float*)d_in[12];
  float* out = (float*)d_out;

  float* ws   = (float*)d_ws;
  float* h    = ws;                            // ROWS*UU
  float* hp   = h    + (size_t)ROWS * UU;      // ROWS*UU
  float* es   = hp   + (size_t)ROWS * UU;      // ROWS
  float* ed   = es   + ROWS;                   // ROWS
  float* mbuf = ed   + ROWS;                   // ROWS
  float* ssum = mbuf + ROWS;                   // ROWS
  float* lg   = ssum + ROWS;                   // BB*EF
  float* agg  = lg   + (size_t)BB * EF;        // ROWS*UU
  float* cbuf = agg  + (size_t)ROWS * UU;      // ROWS*KP
  _Float16* gatF = (_Float16*)(cbuf + (size_t)ROWS * KP);  // 3*4*512
  _Float16* w1F  = gatF + 3 * 4 * 512;                      // 3*8*512
  _Float16* w2F  = w1F  + 3 * 8 * 512;                      // 3*4*512

  const int nEdgeT  = BB * EF;               // 557056
  const int edgeBlk = nEdgeT / 256;          // 2176
  const int aggBlk  = (nEdgeT * 16) / 256;   // 34816
  const int tileBlk = ROWS / 16 / 4;         // 512 (128-thr blocks, wave/tile)
  const int rowBlk  = ROWS / 256;            // 128

  // Pack fragment-ordered fp16 weights (constant across time steps).
  pack_w_kernel<<<(3 * 4 * 512) / 256, 256, 0, stream>>>(gat_W, UU, 4, gatF);
  pack_w_kernel<<<(3 * 8 * 512) / 256, 256, 0, stream>>>(gru1_W, 2 * UU, 8, w1F);
  pack_w_kernel<<<(3 * 4 * 512) / 256, 256, 0, stream>>>(gru2_W, UU, 4, w2F);

  fill_kernel<<<(ROWS * UU) / 256, 256, 0, stream>>>(h, 0.0f, ROWS * UU);

  for (int t = 0; t < TT; ++t) {
    fill_kernel<<<(ROWS * UU) / 256, 256, 0, stream>>>(agg, 0.0f, ROWS * UU);
    fill_kernel<<<rowBlk, 256, 0, stream>>>(mbuf, -INFINITY, ROWS);
    fill_kernel<<<rowBlk, 256, 0, stream>>>(ssum, 0.0f, ROWS);

    gat_gemm_kernel<<<tileBlk, 128, 0, stream>>>(h, x, gatF, hp, t);
    attn_node_kernel<<<rowBlk, 256, 0, stream>>>(hp, a_src, a_dst, es, ed);
    edge_logit_kernel<<<edgeBlk, 256, 0, stream>>>(es, ed, src, dst, mbuf, lg);
    edge_exp_kernel<<<edgeBlk, 256, 0, stream>>>(dst, mbuf, lg, ssum);
    edge_agg_kernel<<<aggBlk, 256, 0, stream>>>(src, dst, lg, ssum, hp, agg);
    build_conc_kernel<<<(ROWS * KP) / 256, 256, 0, stream>>>(x, agg, gat_b,
                                                             cbuf, t);
    gru_kernel<<<tileBlk, 128, 0, stream>>>(cbuf, w1F, w2F, gru1_b, gru2_b, h);
  }
  out_kernel<<<rowBlk, 256, 0, stream>>>(h, out_W, out_b, out);
}
